// ModelLossSemsegGatedCRF_swacorrect_33595234190083
// MI455X (gfx1250) — matrix operand order repaired and checked by reading, not compiled, not run on hardware
//
#include <hip/hip_runtime.h>

typedef __attribute__((ext_vector_type(16))) _Float16 v16h;
typedef __attribute__((ext_vector_type(8)))  _Float16 v8h;
typedef __attribute__((ext_vector_type(8)))  float    v8f;
typedef __attribute__((ext_vector_type(4)))  float    v4f;

#define N_   4
#define C_CH 19
#define H_   128
#define W_   128
#define R_   5
#define HS   512
#define WS   512

// padded channel-last f16 staging buffer: yh[n][row+8][col+8][c:32]
#define PR   144   // 128 rows + 8 halo each side
#define PCW  160   // 128 cols + halo (need -5..+26 beyond x0<=112)
#define PC   32    // channels padded 19 -> 32 (zeros)

// padded pixel-major feature buffer: F[n][row+8][col+8][8]
// c0..2 = 10*rgb (downsampled), c3 = x/6, c4 = y/6, c5..7 = 0; zero halo.
#define FP   144
#define FC   8

// ---------------- Kernel 0: stage y -> padded channel-last f16 -------------
__global__ void crf_stage(const float* __restrict__ y, _Float16* __restrict__ yh) {
  int idx = blockIdx.x * blockDim.x + threadIdx.x;
  const int total = N_ * PR * PCW;
  if (idx >= total) return;
  int pcol = idx % PCW;
  int t    = idx / PCW;
  int prow = t % PR;
  int n    = t / PR;
  int rrow = prow - 8, ccol = pcol - 8;
  bool in = (rrow >= 0) && (rrow < H_) && (ccol >= 0) && (ccol < W_);
  v16h lo = {}, hi = {};
  if (in) {
    const float* yb = y + (size_t)n * C_CH * H_ * W_ + (size_t)rrow * W_ + ccol;
#pragma unroll
    for (int c = 0; c < 16; ++c) lo[c] = (_Float16)yb[(size_t)c * H_ * W_];
#pragma unroll
    for (int c = 16; c < C_CH; ++c) hi[c - 16] = (_Float16)yb[(size_t)c * H_ * W_];
  }
  _Float16* dst = yh + (size_t)idx * PC;
  *(v16h*)dst        = lo;
  *(v16h*)(dst + 16) = hi;
}

// ---------------- Kernel 1: downsample + build padded scaled features ------
__global__ void crf_features(const float* __restrict__ sample,
                             float* __restrict__ fp) {
  int idx = blockIdx.x * blockDim.x + threadIdx.x;
  const int total = N_ * FP * FP;
  if (idx >= total) return;
  int pcol = idx % FP;
  int t    = idx / FP;
  int prow = t % FP;
  int n    = t / FP;
  int rrow = prow - 8, ccol = pcol - 8;
  bool in = (rrow >= 0) && (rrow < H_) && (ccol >= 0) && (ccol < W_);
  v4f f0 = {}, f1 = {};
  if (in) {
#pragma unroll
    for (int ch = 0; ch < 3; ++ch) {
      const float* base = sample + (((size_t)(n * 3 + ch) * HS) + (size_t)rrow * 4) * WS
                          + (size_t)ccol * 4;
      float s = 0.f;
#pragma unroll
      for (int i = 0; i < 4; ++i)
#pragma unroll
        for (int j = 0; j < 4; ++j) s += base[i * WS + j];
      f0[ch] = s * 0.0625f * 10.f;         // rgb / sigma_rgb(0.1)
    }
    f0[3] = (float)ccol * (1.f / 6.f);     // x / sigma_xy(6)
    f1[0] = (float)rrow * (1.f / 6.f);     // y / sigma_xy(6)
  }
  float* dst = fp + (size_t)idx * FC;
  *(v4f*)dst       = f0;
  *(v4f*)(dst + 4) = f1;
}

// ---------------- Kernel 2: main CRF tile kernel (WMMA channel dots) -------
// One wave (32 lanes) per 16-pixel row tile; 4 waves per block.
__launch_bounds__(128)
__global__ void crf_main(const _Float16* __restrict__ yh,
                         const float* __restrict__ fpbuf,
                         const float* __restrict__ cw,
                         float* __restrict__ partials) {
  __shared__ float Dls[4][16][32];   // per-wave 16x26 D tile (padded to 32)
  __shared__ float red[128];

  const int lane = threadIdx.x;      // 0..31
  const int wv   = threadIdx.y;      // 0..3
  const int g    = blockIdx.x * 4 + wv;      // tile id 0..4095
  const int tx   = g & 7;                    // tile column
  const int row  = (g >> 3) & (H_ - 1);
  const int n    = g >> 10;                  // image index
  const int x0   = tx * 16;
  const int m    = lane & 15;
  const int half = lane >> 4;

  const _Float16* ybh = yh    + (size_t)n * PR * PCW * PC;
  const float*    fpb = fpbuf + (size_t)n * FP * FP * FC;

  // ---- A fragment: A[m][c] = y[c, row, x0+m]; channel chunks contiguous in
  // the channel-last layout -> two unconditional b128 loads per lane.
  const int px = x0 + m;
  const _Float16* pp = ybh + ((size_t)(row + 8) * PCW + (size_t)(px + 8)) * PC;
  const v8h a0 = *(const v8h*)(pp + 16 * half);       // channels 16h .. 16h+7
  const v8h a1 = *(const v8h*)(pp + 8 + 16 * half);   // channels 16h+8 .. 16h+15
  v16h A;
#pragma unroll
  for (int e = 0; e < 8; ++e) { A[e] = a0[e]; A[e + 8] = a1[e]; }

  // per-pixel state for the tap phase (lane handles pixel m of the tile)
  const float* fpp = fpb + ((size_t)(row + 8) * FP + (size_t)(px + 8)) * FC;
  const v4f   Pv = *(const v4f*)fpp;
  const float p4 = fpp[4];
  const float cwp = cw[((size_t)n * H_ + row) * W_ + px];

  float accK = 0.f, accS = 0.f;

  for (int dy = -R_; dy <= R_; ++dy) {
    const int qrow = row + dy;

    // ---- B fragments: channel chunk 16*half..+15 contiguous -> one v16h load
    const _Float16* pq = ybh + ((size_t)(qrow + 8) * PCW +
                                (size_t)(x0 - R_ + m + 8)) * PC + 16 * half;
    const v16h B0 = *(const v16h*)pq;                 // neighbor cols x0-5..x0+10
    const v16h B1 = *(const v16h*)(pq + 16 * PC);     // neighbor cols x0+11..x0+26

    // prefetch next dy row band of the staged buffer
    __builtin_prefetch(pq + (size_t)PCW * PC, 0, 1);

    v8f C0 = {}, C1 = {};
    C0 = __builtin_amdgcn_wmma_f32_16x16x32_f16(false, A, false, B0, (short)0, C0, false, false);
    C1 = __builtin_amdgcn_wmma_f32_16x16x32_f16(false, A, false, B1, (short)0, C1, false, false);

    // spill D tiles: f32 C layout — lane gives column, vgpr r gives row r+8*half
#pragma unroll
    for (int r = 0; r < 8; ++r) {
      Dls[wv][r + 8 * half][m]      = C0[r];
      Dls[wv][r + 8 * half][16 + m] = C1[r];
    }
    __syncthreads();

    // ---- tap accumulation: uniform 6-iteration loop, fully branch-free.
    // half 0 -> dx = -5..0 ; half 1 -> dx = 1..6 (dx=6 masked out).
    const int dxb = half ? 1 : -R_;
    const float* fqrow = fpb + ((size_t)(qrow + 8) * FP + (size_t)(px + dxb + 8)) * FC;
#pragma unroll
    for (int j = 0; j < 6; ++j) {
      const int dx = dxb + j;
      const float* fq = fqrow + (size_t)j * FC;
      const v4f   Qv = *(const v4f*)fq;      // zero at padded/OOB -> exact rule
      const float q4 = fq[4];
      const float d0 = Qv[0] - Pv[0];
      const float d1 = Qv[1] - Pv[1];
      const float d2 = Qv[2] - Pv[2];
      const float d3 = Qv[3] - Pv[3];
      const float d4 = q4    - p4;
      const float exy = d3 * d3 + d4 * d4;
      const float e1  = exy + d0 * d0 + d1 * d1 + d2 * d2;
      const float k2  = __expf(-0.5f * exy);
      const float k1  = __expf(-0.5f * e1);
      float keep = 1.f;
      if (half && j == 5) keep = 0.f;                    // phantom tap (dx=6)
      if (!half && j == 5) keep = (dy == 0) ? 0.f : 1.f; // center tap (dx=0)
      const float K = keep * (0.9f * k1 + 0.1f * k2);
      accK += K;
      accS += cwp * K * Dls[wv][m][m + dx + R_];         // D==0 at OOB taps
    }
    __syncthreads();
  }

  // ---- deterministic block reduction of (sumK - S2)
  red[wv * 32 + lane] = accK - accS;
  __syncthreads();
  if (wv == 0 && lane == 0) {
    float s = 0.f;
#pragma unroll 8
    for (int i = 0; i < 128; ++i) s += red[i];
    partials[blockIdx.x] = s;
  }
}

// ---------------- Kernel 3: deterministic final reduction ------------------
__global__ void crf_finalize(const float* __restrict__ partials, int np,
                             float* __restrict__ out) {
  __shared__ float red[256];
  float s = 0.f;
  for (int i = threadIdx.x; i < np; i += 256) s += partials[i];
  red[threadIdx.x] = s;
  __syncthreads();
  if (threadIdx.x == 0) {
    float t = 0.f;
    for (int i = 0; i < 256; ++i) t += red[i];
    out[0] = t / (float)(N_ * H_ * W_);
  }
}

// ---------------------------------------------------------------------------
extern "C" void kernel_launch(void* const* d_in, const int* in_sizes, int n_in,
                              void* d_out, int out_size, void* d_ws, size_t ws_size,
                              hipStream_t stream) {
  const float* y      = (const float*)d_in[0];  // (4,19,128,128)
  const float* sample = (const float*)d_in[1];  // (4,3,512,512)
  const float* cw     = (const float*)d_in[2];  // (4,1,128,128)
  // d_in[3..5]: radius/height/width scalars — shapes fixed, constants baked.

  const size_t yhBytes = (size_t)N_ * PR * PCW * PC * sizeof(_Float16); // 5.9 MB
  const size_t fpBytes = (size_t)N_ * FP * FP * FC * sizeof(float);     // 2.65 MB
  _Float16* yhBuf    = (_Float16*)d_ws;
  float*    fpBuf    = (float*)((char*)d_ws + yhBytes);
  float*    partials = (float*)((char*)d_ws + yhBytes + fpBytes);
  float*    out      = (float*)d_out;

  const int stTotal = N_ * PR * PCW;
  crf_stage<<<(stTotal + 255) / 256, 256, 0, stream>>>(y, yhBuf);

  const int ftTotal = N_ * FP * FP;
  crf_features<<<(ftTotal + 255) / 256, 256, 0, stream>>>(sample, fpBuf);

  dim3 block(32, 4);
  const int nTiles  = N_ * H_ * (W_ / 16);       // 4096 tiles
  const int nBlocks = nTiles / 4;                // 1024 blocks
  crf_main<<<nBlocks, block, 0, stream>>>(yhBuf, fpBuf, cw, partials);

  crf_finalize<<<1, 256, 0, stream>>>(partials, nBlocks, out);
}
